// SimpleHeteroGNN_7086696038636
// MI455X (gfx1250) — matrix operand order repaired and checked by reading, not compiled, not run on hardware
//
#include <hip/hip_runtime.h>
#include <cstddef>

// ---------------- constants ----------------
#define Hn     128
#define Gn     1024
#define OUTn   10
#define BN_EPS 1e-5f

typedef __attribute__((ext_vector_type(8)))  float          v8f;
typedef __attribute__((ext_vector_type(16))) __bf16         v16bf;
typedef __attribute__((ext_vector_type(16))) unsigned short v16u;

static __device__ __forceinline__ unsigned short f2bf(float f) {
    unsigned int u = __float_as_uint(f);
    unsigned int r = u + 0x7FFFu + ((u >> 16) & 1u);   // round-to-nearest-even
    return (unsigned short)(r >> 16);
}
static __device__ __forceinline__ v16bf as_bf(v16u v) {
    union { v16u u; v16bf b; } x; x.u = v; return x.b;
}
static __device__ __forceinline__ unsigned mapf(float f) {
    unsigned b = __float_as_uint(f);
    return (b & 0x80000000u) ? ~b : (b | 0x80000000u);
}
static __device__ __forceinline__ float unmapf(unsigned k) {
    unsigned b = (k & 0x80000000u) ? (k & 0x7FFFFFFFu) : ~k;
    return __uint_as_float(b);
}

// ---------------- tiny utility kernels ----------------
__global__ void zero_f(float* __restrict__ p, int n) {
    int i = blockIdx.x * blockDim.x + threadIdx.x;
    if (i < n) p[i] = 0.f;
}
__global__ void copy_f4(const float4* __restrict__ s, float4* __restrict__ d, int n4) {
    int i = blockIdx.x * blockDim.x + threadIdx.x;
    if (i < n4) d[i] = s[i];
}

// Pack a 128x128 f32 weight (row-major, K x N) into bf16 WMMA B-operand layout:
// dst[((nt*4 + kc)*32 + lane)*16 + j] = W[(kc*32 + (lane>>4)*8 + (j&7) + (j>>3)*16)*128 + nt*16 + (lane&15)]
__global__ void pack_w_bf16(const float* __restrict__ src, unsigned short* __restrict__ dst) {
    int t = blockIdx.x * blockDim.x + threadIdx.x;
    if (t >= Hn * Hn) return;
    int j    = t & 15;
    int lane = (t >> 4) & 31;
    int kc   = (t >> 9) & 3;
    int nt   = t >> 11;
    int k    = kc * 32 + ((lane >> 4) * 8) + (j & 7) + ((j >> 3) * 16);
    int col  = nt * 16 + (lane & 15);
    dst[t] = f2bf(src[(size_t)k * Hn + col]);
}

// ---------------- node encoder ----------------
__global__ void encode_kernel(const int* __restrict__ x, const float* __restrict__ pe,
                              const float* __restrict__ ce, const float* __restrict__ le,
                              float* __restrict__ h, int N) {
    int gid = blockIdx.x * blockDim.x + threadIdx.x;
    int n = gid >> 5;
    if (n >= N) return;
    int lane = gid & 31, c = lane * 4;
    int lv = min(max(x[n * 3 + 0], 0), 999);
    int ca = min(max(x[n * 3 + 1], 0), 4);
    int th = min(max(x[n * 3 + 2], 0), 4999);
    float4 a = *(const float4*)(ce + (size_t)ca * Hn + c);
    float4 b = *(const float4*)(le + (size_t)lv * Hn + c);
    float4 p = *(const float4*)(pe + (size_t)th * Hn + c);
    float4 r; r.x = a.x + b.x + p.x; r.y = a.y + b.y + p.y;
    r.z = a.z + b.z + p.z; r.w = a.w + b.w + p.w;
    *(float4*)(h + (size_t)n * Hn + c) = r;
}

// ---------------- GIN aggregation: agg[dst] += h[src] (wave per edge) ----------------
__global__ void scatter_add(const int* __restrict__ ei, const float* __restrict__ h,
                            float* __restrict__ agg, int E) {
    int gid = blockIdx.x * blockDim.x + threadIdx.x;
    int e = gid >> 5;
    if (e >= E) return;
    int lane = gid & 31;
    int s = ei[e];
    int d = ei[(size_t)E + e];
    float4 v = *(const float4*)(h + (size_t)s * Hn + lane * 4);
    float* dst = agg + (size_t)d * Hn + lane * 4;
    atomicAdd(dst + 0, v.x); atomicAdd(dst + 1, v.y);
    atomicAdd(dst + 2, v.z); atomicAdd(dst + 3, v.w);
}

// ---------------- WMMA GEMM: C[M,128] = f(A[M,128]) @ W[128,128] ----------------
// Compile-time variants: HAS_SS (fused BN affine on A, indices follow A K-layout),
// AREL (relu on transformed A), HAS_BIAS/CREL (epilogue). Bp is pre-swizzled bf16.
template <bool HAS_SS, bool AREL, bool HAS_BIAS, bool CREL>
__global__ void gemm128_wmma(const float* __restrict__ A, const unsigned short* __restrict__ Bp,
                             float* __restrict__ C, const float* __restrict__ ss,
                             const float* __restrict__ bias) {
    int wave  = threadIdx.x >> 5;   // 0..7 -> N-tile
    int lane  = threadIdx.x & 31;
    int m0    = blockIdx.x * 16;
    int n0    = wave * 16;
    int kbase = (lane >> 4) * 8;    // A K-group base
    int col   = n0 + (lane & 15);   // C column
    int mrow  = m0 + (lane & 15);   // A row

    const float* arow = A + (size_t)mrow * Hn;
    const unsigned short* bpw = Bp + ((size_t)(wave * 4) * 32 + lane) * 16;

    v8f c = {};
#pragma unroll
    for (int kc = 0; kc < 4; ++kc) {
        const int kk = kc * 32;
        float va[16];
        {
            float4 p0 = *(const float4*)(arow + kk + kbase + 0);
            float4 p1 = *(const float4*)(arow + kk + kbase + 4);
            float4 p2 = *(const float4*)(arow + kk + kbase + 16);
            float4 p3 = *(const float4*)(arow + kk + kbase + 20);
            va[0]=p0.x;  va[1]=p0.y;  va[2]=p0.z;  va[3]=p0.w;
            va[4]=p1.x;  va[5]=p1.y;  va[6]=p1.z;  va[7]=p1.w;
            va[8]=p2.x;  va[9]=p2.y;  va[10]=p2.z; va[11]=p2.w;
            va[12]=p3.x; va[13]=p3.y; va[14]=p3.z; va[15]=p3.w;
        }
        if (HAS_SS) {
            // per-channel scale/shift, same index pattern as A elements
            float sc[16], sh[16];
            float4 s0 = *(const float4*)(ss + kk + kbase + 0);
            float4 s1 = *(const float4*)(ss + kk + kbase + 4);
            float4 s2 = *(const float4*)(ss + kk + kbase + 16);
            float4 s3 = *(const float4*)(ss + kk + kbase + 20);
            sc[0]=s0.x;  sc[1]=s0.y;  sc[2]=s0.z;  sc[3]=s0.w;
            sc[4]=s1.x;  sc[5]=s1.y;  sc[6]=s1.z;  sc[7]=s1.w;
            sc[8]=s2.x;  sc[9]=s2.y;  sc[10]=s2.z; sc[11]=s2.w;
            sc[12]=s3.x; sc[13]=s3.y; sc[14]=s3.z; sc[15]=s3.w;
            float4 t0 = *(const float4*)(ss + 128 + kk + kbase + 0);
            float4 t1 = *(const float4*)(ss + 128 + kk + kbase + 4);
            float4 t2 = *(const float4*)(ss + 128 + kk + kbase + 16);
            float4 t3 = *(const float4*)(ss + 128 + kk + kbase + 20);
            sh[0]=t0.x;  sh[1]=t0.y;  sh[2]=t0.z;  sh[3]=t0.w;
            sh[4]=t1.x;  sh[5]=t1.y;  sh[6]=t1.z;  sh[7]=t1.w;
            sh[8]=t2.x;  sh[9]=t2.y;  sh[10]=t2.z; sh[11]=t2.w;
            sh[12]=t3.x; sh[13]=t3.y; sh[14]=t3.z; sh[15]=t3.w;
#pragma unroll
            for (int j = 0; j < 16; ++j) va[j] = fmaf(va[j], sc[j], sh[j]);
        }
        // native converts (v_cvt_pk_bf16_f32) instead of integer RNE emulation
        v16bf av;
#pragma unroll
        for (int j = 0; j < 16; ++j) {
            float v = va[j];
            if (AREL) v = fmaxf(v, 0.f);
            av[j] = (__bf16)v;
        }
        v16u bv = *(const v16u*)(bpw + (size_t)kc * 512);   // 32B contiguous fragment
        c = __builtin_amdgcn_wmma_f32_16x16x32_bf16(
                false, av, false, as_bf(bv), (short)0, c, false, false);
    }
    int rbase = (lane >> 4) * 8;
#pragma unroll
    for (int r = 0; r < 8; ++r) {
        int row = m0 + rbase + r;
        float v = c[r];
        if (HAS_BIAS) v += bias[col];
        if (CREL)     v = fmaxf(v, 0.f);
        C[(size_t)row * Hn + col] = v;
    }
}

// ---------------- BatchNorm statistics ----------------
__global__ void colstats(const float* __restrict__ Z, int M, float* __restrict__ stats) {
    int c = threadIdx.x;            // 128 threads
    float s = 0.f, s2 = 0.f;
    for (int r = blockIdx.x; r < M; r += gridDim.x) {
        float v = Z[(size_t)r * Hn + c];
        s += v; s2 += v * v;
    }
    atomicAdd(&stats[c], s);
    atomicAdd(&stats[128 + c], s2);
}
__global__ void bn_finalize(const float* __restrict__ stats, int M,
                            const float* __restrict__ g, const float* __restrict__ be,
                            float* __restrict__ ss) {
    int c = threadIdx.x;
    float inv  = 1.f / (float)M;
    float mean = stats[c] * inv;
    float var  = stats[128 + c] * inv - mean * mean;
    float sc   = g[c] * rsqrtf(var + BN_EPS);
    ss[c]       = sc;
    ss[128 + c] = be[c] - mean * sc;
}
__global__ void bn_relu_apply(const float* __restrict__ Z, const float* __restrict__ ss,
                              float* __restrict__ Y, int total) {
    int i = blockIdx.x * blockDim.x + threadIdx.x;
    if (i >= total) return;
    int c = i & (Hn - 1);
    Y[i] = fmaxf(Z[i] * ss[c] + ss[128 + c], 0.f);
}

// ---------------- gate score (relu'd gate hidden already in Zg) ----------------
__global__ void gate_score(const float* __restrict__ Zg, const float* __restrict__ w2,
                           const float* __restrict__ b2, float* __restrict__ score, int N) {
    int gid = blockIdx.x * blockDim.x + threadIdx.x;
    int n = gid >> 5;
    if (n >= N) return;
    int lane = gid & 31;
    float4 z = *(const float4*)(Zg + (size_t)n * Hn + lane * 4);
    float4 w = *(const float4*)(w2 + lane * 4);
    float acc = z.x * w.x + z.y * w.y + z.z * w.z + z.w * w.w;
    acc += __shfl_xor(acc, 16, 32);
    acc += __shfl_xor(acc, 8, 32);
    acc += __shfl_xor(acc, 4, 32);
    acc += __shfl_xor(acc, 2, 32);
    acc += __shfl_xor(acc, 1, 32);
    if (lane == 0) score[n] = acc + b2[0];
}

// ---------------- segment softmax + attention pooling ----------------
__global__ void seg_max(const float* __restrict__ score, const int* __restrict__ batch,
                        unsigned* __restrict__ gmax, int N) {
    int n = blockIdx.x * blockDim.x + threadIdx.x;
    if (n < N) atomicMax(&gmax[batch[n]], mapf(score[n]));
}
__global__ void seg_expsum(const float* __restrict__ score, const int* __restrict__ batch,
                           const unsigned* __restrict__ gmax, float* __restrict__ gsum,
                           float* __restrict__ ebuf, int N) {
    int n = blockIdx.x * blockDim.x + threadIdx.x;
    if (n >= N) return;
    int b = batch[n];
    float ev = __expf(score[n] - unmapf(gmax[b]));
    ebuf[n] = ev;
    atomicAdd(&gsum[b], ev);
}
__global__ void pool_add(const float* __restrict__ h, const float* __restrict__ ebuf,
                         const float* __restrict__ gsum, const int* __restrict__ batch,
                         float* __restrict__ pooled, int N) {
    int gid = blockIdx.x * blockDim.x + threadIdx.x;
    int n = gid >> 5;
    if (n >= N) return;
    int lane = gid & 31, b = batch[n];
    float alpha = ebuf[n] / gsum[b];
    float4 v = *(const float4*)(h + (size_t)n * Hn + lane * 4);
    float* d = pooled + (size_t)b * Hn + lane * 4;
    atomicAdd(d + 0, alpha * v.x); atomicAdd(d + 1, alpha * v.y);
    atomicAdd(d + 2, alpha * v.z); atomicAdd(d + 3, alpha * v.w);
}

// ---------------- classifier tail: out = relu(BN(zc)) @ W2 + b2 ----------------
__global__ void cls_out(const float* __restrict__ zc, const float* __restrict__ ss,
                        const float* __restrict__ W2, const float* __restrict__ b2,
                        float* __restrict__ out) {
    int t = blockIdx.x * blockDim.x + threadIdx.x;
    if (t >= Gn * OUTn) return;
    int g = t / OUTn, o = t % OUTn;
    float acc = b2[o];
#pragma unroll 4
    for (int c = 0; c < Hn; ++c) {
        float a = fmaxf(zc[(size_t)g * Hn + c] * ss[c] + ss[128 + c], 0.f);
        acc += a * W2[c * OUTn + o];
    }
    out[t] = acc;
}

// ---------------- launch ----------------
extern "C" void kernel_launch(void* const* d_in, const int* in_sizes, int n_in,
                              void* d_out, int out_size, void* d_ws, size_t ws_size,
                              hipStream_t stream) {
    const int N = in_sizes[0] / 3;
    const int E = in_sizes[1] / 2;

    const int*   x      = (const int*)d_in[0];
    const int*   ei     = (const int*)d_in[1];
    const int*   batch  = (const int*)d_in[2];
    const float* pe     = (const float*)d_in[4];
    const float* ce     = (const float*)d_in[5];
    const float* le     = (const float*)d_in[6];
    const float* convW1 = (const float*)d_in[7];
    const float* convG1 = (const float*)d_in[9];
    const float* convBe1= (const float*)d_in[10];
    const float* convW2 = (const float*)d_in[11];
    const float* normG  = (const float*)d_in[13];
    const float* normBe = (const float*)d_in[14];
    const float* gateW1 = (const float*)d_in[15];
    const float* gateB1 = (const float*)d_in[16];
    const float* gateW2 = (const float*)d_in[17];
    const float* gateB2 = (const float*)d_in[18];
    const float* clsW1  = (const float*)d_in[19];
    const float* clsB1  = (const float*)d_in[20];
    const float* clsG   = (const float*)d_in[21];
    const float* clsBe  = (const float*)d_in[22];
    const float* clsW2  = (const float*)d_in[23];
    const float* clsB2  = (const float*)d_in[24];

    // workspace carve-up
    float*          h      = (float*)d_ws;
    float*          agg    = h   + (size_t)N * Hn;
    float*          z      = agg + (size_t)N * Hn;
    unsigned short* wbf    = (unsigned short*)(z + (size_t)N * Hn);   // 8 x 128x128 bf16 (swizzled)
    float*          stats  = (float*)(wbf + 8 * Hn * Hn);
    float*          ss1    = stats + 256;
    float*          ss2    = ss1 + 256;
    float*          score  = ss2 + 256;
    float*          ebuf   = score + N;
    float*          gmax   = ebuf + N;           // used as unsigned
    float*          gsum   = gmax + Gn;
    float*          pooled = gsum + Gn;

    const int TB = 256;
    dim3 blk(TB);
    auto cdiv = [](long long a, long long b) { return (unsigned)((a + b - 1) / b); };

    // bf16 swizzled weight staging: [W1_0,W1_1,W1_2, W2_0,W2_1,W2_2, gateW1, clsW1]
    for (int m = 0; m < 3; ++m)
        pack_w_bf16<<<cdiv(Hn * Hn, TB), blk, 0, stream>>>(convW1 + (size_t)m * Hn * Hn,
                                                           wbf + (size_t)m * Hn * Hn);
    for (int m = 0; m < 3; ++m)
        pack_w_bf16<<<cdiv(Hn * Hn, TB), blk, 0, stream>>>(convW2 + (size_t)m * Hn * Hn,
                                                           wbf + (size_t)(3 + m) * Hn * Hn);
    pack_w_bf16<<<cdiv(Hn * Hn, TB), blk, 0, stream>>>(gateW1, wbf + (size_t)6 * Hn * Hn);
    pack_w_bf16<<<cdiv(Hn * Hn, TB), blk, 0, stream>>>(clsW1,  wbf + (size_t)7 * Hn * Hn);

    // node encoder
    encode_kernel<<<cdiv((long long)N * 32, TB), blk, 0, stream>>>(x, pe, ce, le, h, N);

    // 3 GIN layers
    for (int i = 0; i < 3; ++i) {
        copy_f4<<<cdiv((long long)N * Hn / 4, TB), blk, 0, stream>>>(
            (const float4*)h, (float4*)agg, N * Hn / 4);
        scatter_add<<<cdiv((long long)E * 32, TB), blk, 0, stream>>>(ei, h, agg, E);

        // z = agg @ W1  (conv bias irrelevant before BN)
        gemm128_wmma<false, false, false, false><<<N / 16, blk, 0, stream>>>(
            agg, wbf + (size_t)i * Hn * Hn, z, nullptr, nullptr);
        zero_f<<<1, 256, 0, stream>>>(stats, 256);
        colstats<<<1024, 128, 0, stream>>>(z, N, stats);
        bn_finalize<<<1, 128, 0, stream>>>(stats, N, convG1 + i * Hn, convBe1 + i * Hn, ss1);

        // agg = relu(BN(z)) @ W2   (BN+relu fused into A load)
        gemm128_wmma<true, true, false, false><<<N / 16, blk, 0, stream>>>(
            z, wbf + (size_t)(3 + i) * Hn * Hn, agg, ss1, nullptr);
        zero_f<<<1, 256, 0, stream>>>(stats, 256);
        colstats<<<1024, 128, 0, stream>>>(agg, N, stats);
        bn_finalize<<<1, 128, 0, stream>>>(stats, N, normG + i * Hn, normBe + i * Hn, ss2);
        bn_relu_apply<<<cdiv((long long)N * Hn, TB), blk, 0, stream>>>(agg, ss2, h, N * Hn);
    }

    // gate hidden: z = relu(h @ gateW1 + b1)
    gemm128_wmma<false, false, true, true><<<N / 16, blk, 0, stream>>>(
        h, wbf + (size_t)6 * Hn * Hn, z, nullptr, gateB1);
    gate_score<<<cdiv((long long)N * 32, TB), blk, 0, stream>>>(z, gateW2, gateB2, score, N);

    // zero gmax|gsum|pooled (contiguous: Gn + Gn + Gn*128 floats)
    zero_f<<<cdiv(Gn * (Hn + 2), TB), blk, 0, stream>>>(gmax, Gn * (Hn + 2));
    seg_max<<<cdiv(N, TB), blk, 0, stream>>>(score, batch, (unsigned*)gmax, N);
    seg_expsum<<<cdiv(N, TB), blk, 0, stream>>>(score, batch, (const unsigned*)gmax, gsum, ebuf, N);
    pool_add<<<cdiv((long long)N * 32, TB), blk, 0, stream>>>(h, ebuf, gsum, batch, pooled, N);

    // classifier: zc = pooled @ clsW1 + b1 ; BN stats over G rows ; out
    gemm128_wmma<false, false, true, false><<<Gn / 16, blk, 0, stream>>>(
        pooled, wbf + (size_t)7 * Hn * Hn, z, nullptr, clsB1);
    zero_f<<<1, 256, 0, stream>>>(stats, 256);
    colstats<<<256, 128, 0, stream>>>(z, Gn, stats);
    bn_finalize<<<1, 128, 0, stream>>>(stats, Gn, clsG, clsBe, ss1);
    cls_out<<<cdiv(Gn * OUTn, TB), blk, 0, stream>>>(z, ss1, clsW2, clsB2, (float*)d_out);
}
